// SetFlowUpsampling_88201448391140
// MI455X (gfx1250) — matrix operand order repaired and checked by reading, not compiled, not run on hardware
//
#include <hip/hip_runtime.h>
#include <hip/hip_bf16.h>

typedef __attribute__((ext_vector_type(16))) _Float16 v16h;
typedef __attribute__((ext_vector_type(8)))  float    v8f;

#define B_   2
#define N1_  4096
#define N2_  16384
#define C_   64
#define K1_  67      // C + 3
#define NS_  8       // neighbors
#define NCOL (B_ * N2_ * NS_)       // 262144 GEMM columns
#define NTILE (NCOL / 16)           // 16384 tiles of 16 columns

// ---------------------------------------------------------------------------
// Kernel 1: KNN — one thread per query, xyz1 staged in LDS (48 KB), branchless
// 8-deep insertion network with early-out guard.
// ---------------------------------------------------------------------------
__global__ void knn_kernel(const float* __restrict__ xyz1,
                           const float* __restrict__ xyz2,
                           int* __restrict__ idxout) {
    __shared__ float sx[N1_], sy[N1_], sz[N1_];
    const int b  = blockIdx.x >> 6;                       // 64 blocks per batch
    const int n2 = ((blockIdx.x & 63) << 8) + threadIdx.x;

    for (int i = threadIdx.x; i < N1_; i += 256) {
        sx[i] = xyz1[(b * 3 + 0) * N1_ + i];
        sy[i] = xyz1[(b * 3 + 1) * N1_ + i];
        sz[i] = xyz1[(b * 3 + 2) * N1_ + i];
    }
    __syncthreads();

    const float qx = xyz2[(b * 3 + 0) * N2_ + n2];
    const float qy = xyz2[(b * 3 + 1) * N2_ + n2];
    const float qz = xyz2[(b * 3 + 2) * N2_ + n2];

    float bd[NS_]; int bi[NS_];
#pragma unroll
    for (int i = 0; i < NS_; ++i) { bd[i] = 3.4e38f; bi[i] = 0; }

    for (int j = 0; j < N1_; ++j) {
        const float dx = sx[j] - qx;
        const float dy = sy[j] - qy;
        const float dz = sz[j] - qz;
        float d = fmaf(dx, dx, fmaf(dy, dy, dz * dz));
        if (d < bd[NS_ - 1]) {             // early-out: insertion is rare
            float cd = d; int ci = j;
#pragma unroll
            for (int i = 0; i < NS_; ++i) {  // keep ascending; stable on ties
                const bool sw = cd < bd[i];
                const float td = bd[i]; const int ti = bi[i];
                bd[i] = sw ? cd : td;  bi[i] = sw ? ci : ti;
                cd    = sw ? td : cd;  ci    = sw ? ti : ci;
            }
        }
    }
    const long base = ((long)b * N2_ + n2) * NS_;
#pragma unroll
    for (int i = 0; i < NS_; ++i) idxout[base + i] = bi[i];
}

// ---------------------------------------------------------------------------
// Kernel 2: repack feat1 [B,C,N1] (f32) -> featT [B,N1,C] (f16), coalesced
// writes so each point's 64 channels are contiguous for vector B-frag loads.
// ---------------------------------------------------------------------------
__global__ void transpose_feat(const float* __restrict__ feat1,
                               _Float16* __restrict__ featT) {
    const int i  = blockIdx.x * blockDim.x + threadIdx.x; // over B*N1*C
    const int c  = i & (C_ - 1);
    const int n1 = (i >> 6) & (N1_ - 1);
    const int b  = i >> 18;
    featT[i] = (_Float16)feat1[((b * C_ + c) << 12) + n1];
}

// ---------------------------------------------------------------------------
// Kernel 3: gathered MLP via WMMA. One wave per 16-column tile
// (16 columns = 2 query points x 8 neighbors).
//   feat part (K=0..63):  8x v_wmma_f32_16x16x32_f16 (A register-resident)
//   dxyz part (K=64..66): f32 VALU FMAs with weights from LDS
// then bias + LeakyReLU, W2 (LDS), softmax over 8-lane groups, flow sum.
// ---------------------------------------------------------------------------
__global__ void __launch_bounds__(256, 1)
mlp_kernel(const float* __restrict__ xyz1,
           const float* __restrict__ xyz2,
           const float* __restrict__ flow,
           const float* __restrict__ W1,
           const float* __restrict__ b1,
           const float* __restrict__ W2,
           const float* __restrict__ b2,
           const int* __restrict__ idx,
           const _Float16* __restrict__ featT,
           float* __restrict__ out) {
    __shared__ float sW2[3 * C_];   // W2[o][r]
    __shared__ float sW1x[3 * C_];  // W1[r][64+o] (dxyz columns)
    __shared__ float sb1[C_];
    __shared__ float sb2[3];
    const int tid = threadIdx.x;
    if (tid < 3 * C_) sW2[tid] = W2[tid];
    if (tid < 3 * C_) {             // tid = o*64 + r
        const int o = tid >> 6, r = tid & 63;
        sW1x[tid] = W1[r * K1_ + C_ + o];
    }
    if (tid < C_) sb1[tid] = b1[tid];
    if (tid < 3)  sb2[tid] = b2[tid];
    __syncthreads();

    const int lane = tid & 31;
    const int wave = tid >> 5;
    const int wavesPerBlock = blockDim.x >> 5;
    const int m  = lane & 15;       // row (A) / column (B,D) within tile
    const int hi = lane >> 4;       // which K/M half this lane holds

    // ---- A fragments: W1 feat columns only (K=0..63), 4 M-tiles x 2 K-tiles.
    // A in this code's contiguous k order (same as B): lane holds M = lane&15, K = hi*16 + e.
    v16h Afrag[8];
#pragma unroll
    for (int mt = 0; mt < 4; ++mt) {
        const int r = mt * 16 + m;
#pragma unroll
        for (int kt = 0; kt < 2; ++kt) {
            v16h a;
#pragma unroll
            for (int e = 0; e < 16; ++e) {
                const int k = kt * 32 + hi * 16 + e;
                a[e] = (_Float16)W1[r * K1_ + k];
            }
            Afrag[mt * 2 + kt] = a;
        }
    }

    for (int tile = blockIdx.x * wavesPerBlock + wave; tile < NTILE;
         tile += gridDim.x * wavesPerBlock) {
        const int p    = tile * 16 + m;          // global GEMM column
        const int bidx = p >> 17;                // / (N2*8)
        const int rem  = p & (N2_ * NS_ - 1);
        const int n2   = rem >> 3;
        const int n1   = idx[p];

        v8f acc[4];
#pragma unroll
        for (int mt = 0; mt < 4; ++mt) acc[mt] = {};

        // ---- K-tiles 0,1: gathered features. B in contiguous k order (matching A): lane holds
        // column m, K = hi*16 + e  ==> 16 consecutive channels = one v16h load.
        const _Float16* col = featT + ((size_t)(bidx * N1_ + n1)) * C_;
#pragma unroll
        for (int kt = 0; kt < 2; ++kt) {
            const v16h bfr = *(const v16h*)(col + kt * 32 + hi * 16);
#pragma unroll
            for (int mt = 0; mt < 4; ++mt)
                acc[mt] = __builtin_amdgcn_wmma_f32_16x16x32_f16(
                    false, Afrag[mt * 2 + kt], false, bfr,
                    (short)0, acc[mt], false, false);
        }

        // ---- dxyz contribution (K=64..66) in f32 on the VALU, plus bias,
        // LeakyReLU(0.1) and W2 partials over this lane's 32 rows.
        // D layout: VGPR v holds row r = mt*16 + hi*8 + v, column m.
        const float dx = xyz1[(bidx * 3 + 0) * N1_ + n1] - xyz2[(bidx * 3 + 0) * N2_ + n2];
        const float dy = xyz1[(bidx * 3 + 1) * N1_ + n1] - xyz2[(bidx * 3 + 1) * N2_ + n2];
        const float dz = xyz1[(bidx * 3 + 2) * N1_ + n1] - xyz2[(bidx * 3 + 2) * N2_ + n2];

        float s0 = 0.f, s1 = 0.f, s2 = 0.f;
#pragma unroll
        for (int mt = 0; mt < 4; ++mt) {
#pragma unroll
            for (int v = 0; v < 8; ++v) {
                const int r = mt * 16 + hi * 8 + v;
                float h = acc[mt][v] + sb1[r];
                h = fmaf(sW1x[0 * C_ + r], dx, h);
                h = fmaf(sW1x[1 * C_ + r], dy, h);
                h = fmaf(sW1x[2 * C_ + r], dz, h);
                h = (h >= 0.f) ? h : 0.1f * h;
                s0 = fmaf(sW2[0 * C_ + r], h, s0);
                s1 = fmaf(sW2[1 * C_ + r], h, s1);
                s2 = fmaf(sW2[2 * C_ + r], h, s2);
            }
        }
        // lane l and l^16 hold the two row-halves of the same column
        s0 += __shfl_xor(s0, 16, 32);
        s1 += __shfl_xor(s1, 16, 32);
        s2 += __shfl_xor(s2, 16, 32);
        float w0 = s0 + sb2[0], w1 = s1 + sb2[1], w2 = s2 + sb2[2];

        // ---- softmax over the 8 neighbors (8-lane groups: xor 1,2,4)
        float m0 = w0, m1 = w1, m2 = w2;
#pragma unroll
        for (int msk = 1; msk <= 4; msk <<= 1) {
            m0 = fmaxf(m0, __shfl_xor(m0, msk, 32));
            m1 = fmaxf(m1, __shfl_xor(m1, msk, 32));
            m2 = fmaxf(m2, __shfl_xor(m2, msk, 32));
        }
        float e0 = __expf(w0 - m0), e1 = __expf(w1 - m1), e2 = __expf(w2 - m2);
        float t0 = e0, t1 = e1, t2 = e2;
#pragma unroll
        for (int msk = 1; msk <= 4; msk <<= 1) {
            t0 += __shfl_xor(t0, msk, 32);
            t1 += __shfl_xor(t1, msk, 32);
            t2 += __shfl_xor(t2, msk, 32);
        }

        // ---- weighted flow sum over neighbors
        float f0 = flow[(bidx * 3 + 0) * N1_ + n1] * (e0 / t0);
        float f1 = flow[(bidx * 3 + 1) * N1_ + n1] * (e1 / t1);
        float f2 = flow[(bidx * 3 + 2) * N1_ + n1] * (e2 / t2);
#pragma unroll
        for (int msk = 1; msk <= 4; msk <<= 1) {
            f0 += __shfl_xor(f0, msk, 32);
            f1 += __shfl_xor(f1, msk, 32);
            f2 += __shfl_xor(f2, msk, 32);
        }
        if (hi == 0 && (lane & 7) == 0) {      // lanes 0 and 8: the 2 n2's
            out[(bidx * 3 + 0) * N2_ + n2] = f0;
            out[(bidx * 3 + 1) * N2_ + n2] = f1;
            out[(bidx * 3 + 2) * N2_ + n2] = f2;
        }
    }
}

// ---------------------------------------------------------------------------
extern "C" void kernel_launch(void* const* d_in, const int* in_sizes, int n_in,
                              void* d_out, int out_size, void* d_ws, size_t ws_size,
                              hipStream_t stream) {
    (void)in_sizes; (void)n_in; (void)out_size; (void)ws_size;
    const float* xyz1  = (const float*)d_in[0];
    const float* xyz2  = (const float*)d_in[1];
    const float* feat1 = (const float*)d_in[2];
    const float* flow  = (const float*)d_in[3];
    const float* W1    = (const float*)d_in[4];
    const float* b1    = (const float*)d_in[5];
    const float* W2    = (const float*)d_in[6];
    const float* b2    = (const float*)d_in[7];
    float* out = (float*)d_out;

    int*      ws_idx = (int*)d_ws;                                  // 1 MB
    _Float16* featT  = (_Float16*)((char*)d_ws + (size_t)NCOL * 4); // 1 MB

    knn_kernel<<<(B_ * N2_) / 256, 256, 0, stream>>>(xyz1, xyz2, ws_idx);
    transpose_feat<<<(B_ * N1_ * C_) / 256, 256, 0, stream>>>(feat1, featT);
    mlp_kernel<<<512, 256, 0, stream>>>(xyz1, xyz2, flow, W1, b1, W2, b2,
                                        ws_idx, featT, out);
}
